// HMLSTMCell1_6657199309450
// MI455X (gfx1250) — matrix operand order repaired and checked by reading, not compiled, not run on hardware
//
#include <hip/hip_runtime.h>
#include <math.h>

#define B_ROWS 8192
#define HID    1024
#define K3     3072   // concatenated K: IBS + H + ITS
#define G4     4096   // 4*H
#define GW     4097   // weight row width (G + sz column)

#define BM   128
#define BN   32
#define BK   32
#define LSTR 40       // LDS row stride in bf16 (16B-aligned rows, conflict-dodging pad)

typedef __bf16 bf16_t;
typedef __attribute__((ext_vector_type(16))) __bf16 v16bf;
typedef __attribute__((ext_vector_type(8)))  __bf16 v8bf;
typedef __attribute__((ext_vector_type(8)))  float  v8f;
typedef int v4i __attribute__((vector_size(16)));   // matches builtin param type

typedef __attribute__((address_space(1))) v4i* gptr_v4i;
typedef __attribute__((address_space(3))) v4i* lptr_v4i;

__device__ __forceinline__ bf16_t f2bf(float f) {
  unsigned u = __builtin_bit_cast(unsigned, f);
  u += 0x7FFFu + ((u >> 16) & 1u);            // round-to-nearest-even
  unsigned short h = (unsigned short)(u >> 16);
  return __builtin_bit_cast(bf16_t, h);
}

__device__ __forceinline__ float sigm(float x) { return 1.0f / (1.0f + __expf(-x)); }

// ---- CDNA5 async global->LDS copy (ASYNCcnt-tracked), 16 bytes ------------
__device__ __forceinline__ unsigned lds_byte_off(const void* p) {
  return (unsigned)(size_t)(__attribute__((address_space(3))) const char*)p;
}

__device__ __forceinline__ void async_copy_b128(const bf16_t* gsrc, bf16_t* ldst) {
#if __has_builtin(__builtin_amdgcn_global_load_async_to_lds_b128)
  __builtin_amdgcn_global_load_async_to_lds_b128(
      (gptr_v4i)gsrc, (lptr_v4i)ldst, 0, 0);
#else
  asm volatile("global_load_async_to_lds_b128 %0, %1, off"
               :: "v"(lds_byte_off(ldst)), "v"(gsrc) : "memory");
#endif
}

__device__ __forceinline__ void wait_async0() {
#if __has_builtin(__builtin_amdgcn_s_wait_asynccnt)
  __builtin_amdgcn_s_wait_asynccnt(0);
#else
  asm volatile("s_wait_asynccnt 0x0" ::: "memory");
#endif
}

// ---------------------------------------------------------------------------
// Prep 1: Acat[r, 0:3072] = bf16([hb | h | ht]) with hb = h_bottom*z_bottom,
//         ht = h_top*z.  Row-major, stride K3.
// ---------------------------------------------------------------------------
__global__ __launch_bounds__(256)
void prep_acat(const float* __restrict__ hb, const float* __restrict__ h,
               const float* __restrict__ ht, const float* __restrict__ z,
               const float* __restrict__ zb, bf16_t* __restrict__ A) {
  size_t idx = (size_t)blockIdx.x * blockDim.x + threadIdx.x;  // < B_ROWS*K3
  int r   = (int)(idx / K3);
  int col = (int)(idx % K3);
  float v;
  if (col < HID)           v = hb[(size_t)r * HID + col] * zb[r];
  else if (col < 2 * HID)  v = h [(size_t)r * HID + (col - HID)];
  else                     v = ht[(size_t)r * HID + (col - 2 * HID)] * z[r];
  A[idx] = f2bf(v);
}

// ---------------------------------------------------------------------------
// Prep 2: Bt[j, k] = bf16 of [W;R;U][k, j]  (concatenated weights, transposed)
// ---------------------------------------------------------------------------
__global__ __launch_bounds__(256)
void prep_bt(const float* __restrict__ W, const float* __restrict__ R,
             const float* __restrict__ U, bf16_t* __restrict__ Bt) {
  size_t idx = (size_t)blockIdx.x * blockDim.x + threadIdx.x;  // < G4*K3
  int j = (int)(idx / K3);
  int k = (int)(idx % K3);
  float v;
  if (k < HID)            v = W[(size_t)k * GW + j];
  else if (k < 2 * HID)   v = R[(size_t)(k - HID) * GW + j];
  else                    v = U[(size_t)(k - 2 * HID) * GW + j];
  Bt[idx] = f2bf(v);
}

// ---------------------------------------------------------------------------
// Main fused kernel: double-buffered async-LDS bf16 WMMA GEMM over K=3072,
// per block a 128x32 tile of hidden columns for ALL FOUR gates, then the
// HM-LSTM elementwise epilogue fully in registers -> h_new, c_new.
// ---------------------------------------------------------------------------
__global__ __launch_bounds__(256)
void hmlstm_gemm(const bf16_t* __restrict__ A, const bf16_t* __restrict__ Bt,
                 const float* __restrict__ bias, const float* __restrict__ z,
                 const float* __restrict__ zb, const float* __restrict__ c,
                 const float* __restrict__ h, float* __restrict__ out) {
  __shared__ __align__(16) bf16_t As[2][BM * LSTR];    // 2 x 10 KB
  __shared__ __align__(16) bf16_t Bs[2][128 * LSTR];   // 2 x 10 KB  (B^T tiles)

  const int tid  = threadIdx.x;
  const int wid  = tid >> 5;        // wave id 0..7 -> 16-row strip
  const int lane = tid & 31;
  const int half = lane >> 4;       // lane group 0/1
  const int nIn  = lane & 15;       // row (A frag) / column (B frag) / N (C frag)

  const int rowBase = blockIdx.y * BM;
  const int nBase   = blockIdx.x * BN;

  // cooperative fill mapping: 2 threads per row, 16 bf16 (two B128) each
  const int lr = tid >> 1;            // 0..127
  const int kh = (tid & 1) * 16;      // 0 or 16
  const int gate_ld = lr >> 5;        // which gate chunk for B
  const int nn_ld   = lr & 31;        // column within gate tile

  const bf16_t* gA = A  + (size_t)(rowBase + lr) * K3 + kh;
  const bf16_t* gB = Bt + (size_t)(gate_ld * HID + nBase + nn_ld) * K3 + kh;

  v8f acc[4][2];
#pragma unroll
  for (int t = 0; t < 4; ++t)
#pragma unroll
    for (int nt = 0; nt < 2; ++nt)
#pragma unroll
      for (int e = 0; e < 8; ++e) acc[t][nt][e] = 0.0f;

  // prologue: async-load tile 0 into buffer 0
  async_copy_b128(gA,     &As[0][lr * LSTR + kh]);
  async_copy_b128(gA + 8, &As[0][lr * LSTR + kh + 8]);
  async_copy_b128(gB,     &Bs[0][lr * LSTR + kh]);
  async_copy_b128(gB + 8, &Bs[0][lr * LSTR + kh + 8]);
  wait_async0();
  __syncthreads();

  int buf = 0;
  for (int k0 = 0; k0 < K3; k0 += BK) {
    // kick off async DMA for the next tile while we compute on this one
    if (k0 + BK < K3) {
      const int nb = buf ^ 1;
      const bf16_t* nA = gA + (k0 + BK);
      const bf16_t* nB = gB + (k0 + BK);
      async_copy_b128(nA,     &As[nb][lr * LSTR + kh]);
      async_copy_b128(nA + 8, &As[nb][lr * LSTR + kh + 8]);
      async_copy_b128(nB,     &Bs[nb][lr * LSTR + kh]);
      async_copy_b128(nB + 8, &Bs[nb][lr * LSTR + kh + 8]);
    }

    // A fragment (16x32): lane holds row nIn; K = {half*8..+7, 16+half*8..+7}
    v16bf afr;
    {
      const bf16_t* ap = &As[buf][(wid * 16 + nIn) * LSTR];
      ((v8bf*)&afr)[0] = *(const v8bf*)(ap + half * 8);
      ((v8bf*)&afr)[1] = *(const v8bf*)(ap + 16 + half * 8);
    }
    // preload all 8 B fragments, then chain the WMMAs back-to-back
    v16bf bfr[8];
#pragma unroll
    for (int i = 0; i < 8; ++i) {
      const bf16_t* bp = &Bs[buf][(i * 16 + nIn) * LSTR + half * 8];
      ((v8bf*)&bfr[i])[0] = *(const v8bf*)(bp);
      ((v8bf*)&bfr[i])[1] = *(const v8bf*)(bp + 16);
    }
#pragma unroll
    for (int t = 0; t < 4; ++t)
#pragma unroll
      for (int nt = 0; nt < 2; ++nt)
        acc[t][nt] = __builtin_amdgcn_wmma_f32_16x16x32_bf16(
            false, afr, false, bfr[t * 2 + nt], (short)0, acc[t][nt], false, false);

    wait_async0();       // my DMA into the next buffer has landed
    __syncthreads();     // everyone's DMA landed; everyone done reading `buf`
    buf ^= 1;
  }

  // ---- fused HM-LSTM epilogue (C layout: elem v -> M = v + 8*half, N = nIn)
#pragma unroll
  for (int nt = 0; nt < 2; ++nt) {
    const int j  = nBase + nt * 16 + nIn;
    const float bi  = bias[j];
    const float bg  = bias[HID + j];
    const float bo  = bias[2 * HID + j];
    const float bff = bias[3 * HID + j];
#pragma unroll
    for (int v = 0; v < 8; ++v) {
      const int r = rowBase + wid * 16 + half * 8 + v;
      const float zr = z[r], zbr = zb[r];
      const bool flush  = (zr == 1.0f);
      const bool update = (!flush) && (zbr == 1.0f);
      const size_t off = (size_t)r * HID + j;
      const float i_ = sigm(acc[0][nt][v] + bi);
      const float g_ = tanhf(acc[1][nt][v] + bg);
      const float o_ = sigm(acc[2][nt][v] + bo);
      const float f_ = sigm(acc[3][nt][v] + bff);
      const float cv = c[off];
      const float ig = i_ * g_;
      const float cn = flush ? ig : (update ? cv * f_ + ig : cv);
      const float hn = (flush || update) ? tanhf(cn) * o_ : h[off];
      out[off] = hn;                              // h_new
      out[(size_t)B_ROWS * HID + off] = cn;       // c_new
    }
  }
}

// ---------------------------------------------------------------------------
// Boundary detector: sz = sigmoid(hb.W[:,G] + h_new.R[:,G] + ht.U[:,G] + b[G])
// z_out = hard-threshold. One wave per batch row.
// ---------------------------------------------------------------------------
__global__ __launch_bounds__(256)
void hmlstm_z(const float* __restrict__ hb0, const float* __restrict__ ht0,
              const float* __restrict__ z, const float* __restrict__ zb,
              const float* __restrict__ W, const float* __restrict__ R,
              const float* __restrict__ U, const float* __restrict__ bias,
              float* __restrict__ out) {
  const int wid = threadIdx.x >> 5, lane = threadIdx.x & 31;
  const int r = blockIdx.x * 8 + wid;
  const float* hnew = out;   // written by hmlstm_gemm earlier on this stream
  const float zr = z[r], zbr = zb[r];
  float acc = 0.0f;
  for (int k = lane; k < HID; k += 32) {
    acc += (hb0[(size_t)r * HID + k] * zbr) * W[(size_t)k * GW + G4];
    acc +=  hnew[(size_t)r * HID + k]       * R[(size_t)k * GW + G4];
    acc += (ht0[(size_t)r * HID + k] * zr)  * U[(size_t)k * GW + G4];
  }
#pragma unroll
  for (int off = 16; off > 0; off >>= 1) acc += __shfl_down(acc, off, 32);
  if (lane == 0) {
    const float s4 = acc + bias[G4];
    const float sz = sigm(s4);
    float zt = 0.5f * (sz + 1.0f);
    zt = fminf(fmaxf(zt, 0.0f), 1.0f);
    out[(size_t)2 * B_ROWS * HID + r] = (zt > 0.5f) ? 1.0f : 0.0f;
  }
}

// ---------------------------------------------------------------------------
extern "C" void kernel_launch(void* const* d_in, const int* in_sizes, int n_in,
                              void* d_out, int out_size, void* d_ws, size_t ws_size,
                              hipStream_t stream) {
  const float* h_bottom = (const float*)d_in[0];
  const float* h        = (const float*)d_in[1];
  const float* h_top    = (const float*)d_in[2];
  const float* c        = (const float*)d_in[3];
  const float* z        = (const float*)d_in[4];
  const float* zb       = (const float*)d_in[5];
  const float* W        = (const float*)d_in[6];
  const float* R        = (const float*)d_in[7];
  const float* U        = (const float*)d_in[8];
  const float* b        = (const float*)d_in[9];
  float* out = (float*)d_out;

  bf16_t* Acat = (bf16_t*)d_ws;                                        // 8192*3072*2 B
  bf16_t* Btw  = (bf16_t*)((char*)d_ws + (size_t)B_ROWS * K3 * 2);     // 4096*3072*2 B

  prep_acat<<<(unsigned)(((size_t)B_ROWS * K3) / 256), 256, 0, stream>>>(
      h_bottom, h, h_top, z, zb, Acat);
  prep_bt<<<(unsigned)(((size_t)G4 * K3) / 256), 256, 0, stream>>>(W, R, U, Btw);

  dim3 grid(HID / BN, B_ROWS / BM);  // 32 x 64 blocks
  hmlstm_gemm<<<grid, 256, 0, stream>>>(Acat, Btw, b, z, zb, c, h, out);

  hmlstm_z<<<B_ROWS / 8, 256, 0, stream>>>(h_bottom, h_top, z, zb, W, R, U, b, out);
}
